// SAN_47124381172162
// MI455X (gfx1250) — compile-verified, matrix operand order checked by reading, not run on hardware
//
#include <hip/hip_runtime.h>

typedef __attribute__((ext_vector_type(16))) _Float16 v16h;
typedef __attribute__((ext_vector_type(8)))  float    v8f;
typedef __attribute__((ext_vector_type(4)))  int      v4i;

#define NB   8
#define NC   256
#define NH   128
#define NHW  (NH * NH)        /* 16384 */
#define HM   512
#define EPSV 1e-5f
#define SCL  5.0f

// gfx1250 async global->LDS path (ASYNCcnt), guarded so compile never regresses.
#if defined(__has_builtin)
#  if __has_builtin(__builtin_amdgcn_global_load_async_to_lds_b128) && \
      __has_builtin(__builtin_amdgcn_s_wait_asynccnt)
#    define USE_ASYNC_LDS 1
#  endif
#endif
#ifndef USE_ASYNC_LDS
#  define USE_ASYNC_LDS 0
#endif

// Builtin signature (from hipcc diagnostic): param0 = v4i* (global src),
// param1 = v4i* (LDS dst), then imm offset, imm cpol.
#define AS_GLOBAL_V4I(p) ((__attribute__((address_space(1))) v4i*)(void*)(p))
#define AS_LDS_V4I(p)    ((__attribute__((address_space(3))) v4i*)(void*)(p))

// ---------------------------------------------------------------------------
// K1: antialiased bilinear downsample masks [8,512,512] -> m [8,128,128]
// JAX resize (antialias=True), scale=4: taps j = 4*i-2 .. 4*i+5,
// raw weights {1,3,5,7,7,5,3,1}/32, per-dim renormalization at edges.
// ---------------------------------------------------------------------------
__global__ void k_resize(const float* __restrict__ masks, float* __restrict__ m) {
    int tid = blockIdx.x * blockDim.x + threadIdx.x;   // exactly 8*16384 threads
    int b = tid / NHW, p = tid % NHW;
    int y = p / NH, x = p % NH;
    const float WT[8] = {1.f/32.f, 3.f/32.f, 5.f/32.f, 7.f/32.f,
                         7.f/32.f, 5.f/32.f, 3.f/32.f, 1.f/32.f};
    const float* mb = masks + (size_t)b * HM * HM;
    float acc = 0.f, wsy = 0.f, wsx = 0.f;
#pragma unroll
    for (int tx = 0; tx < 8; ++tx) {
        int jx = 4 * x - 2 + tx;
        if (jx >= 0 && jx < HM) wsx += WT[tx];
    }
#pragma unroll
    for (int ty = 0; ty < 8; ++ty) {
        int jy = 4 * y - 2 + ty;
        if (jy < 0 || jy >= HM) continue;
        wsy += WT[ty];
        float rowacc = 0.f;
        const float* row = mb + (size_t)jy * HM;
#pragma unroll
        for (int tx = 0; tx < 8; ++tx) {
            int jx = 4 * x - 2 + tx;
            if (jx >= 0 && jx < HM) rowacc += WT[tx] * row[jx];
        }
        acc += WT[ty] * rowacc;
    }
    m[tid] = acc / (wsy * wsx);
}

// ---------------------------------------------------------------------------
// K2: per-pixel channel mean & max of mid = x*m.  One thread per (b,pixel),
// strided walk over 256 channels (coalesced across threads each iteration).
// ---------------------------------------------------------------------------
__global__ void k_chanreduce(const float* __restrict__ x, const float* __restrict__ m,
                             float* __restrict__ avg, float* __restrict__ mx) {
    int tid = blockIdx.x * blockDim.x + threadIdx.x;   // exactly 8*16384 threads
    int b = tid / NHW, p = tid % NHW;
    float mv = m[tid];
    const float* xp = x + (size_t)b * NC * NHW + p;
    float s = 0.f, mxv = -3.402823466e38f;
    for (int c = 0; c < NC; ++c) {
        if (c + 8 < NC) __builtin_prefetch(xp + (size_t)(c + 8) * NHW, 0, 0);
        float v = xp[(size_t)c * NHW] * mv;
        s += v;
        mxv = fmaxf(mxv, v);
    }
    avg[tid] = s * (1.f / (float)NC);
    mx[tid]  = mxv;
}

// ---------------------------------------------------------------------------
// K3: 7x7 conv over [avg, max, m] (zero pad 3) + sigmoid; emit gate g = m*sig.
// ---------------------------------------------------------------------------
__global__ void k_conv_gate(const float* __restrict__ avg, const float* __restrict__ mx,
                            const float* __restrict__ m, const float* __restrict__ w,
                            float* __restrict__ g) {
    __shared__ float sw[147];
    if (threadIdx.x < 147) sw[threadIdx.x] = w[threadIdx.x];
    __syncthreads();
    int tid = blockIdx.x * blockDim.x + threadIdx.x;   // exactly 8*16384 threads
    int b = tid / NHW, p = tid % NHW;
    int y = p / NH, x = p % NH;
    float acc = 0.f;
#pragma unroll
    for (int ky = 0; ky < 7; ++ky) {
        int iy = y + ky - 3;
        if (iy < 0 || iy >= NH) continue;
#pragma unroll
        for (int kx = 0; kx < 7; ++kx) {
            int ix = x + kx - 3;
            if (ix < 0 || ix >= NH) continue;
            int q = b * NHW + iy * NH + ix;
            int t = ky * 7 + kx;
            acc += sw[t] * avg[q] + sw[49 + t] * mx[q] + sw[98 + t] * m[q];
        }
    }
    float sig = 1.f / (1.f + __expf(-acc));
    g[tid] = m[tid] * sig;
}

// ---------------------------------------------------------------------------
// K4: InstanceNorm statistics via WMMA reduction.  One block per (b,c);
// each wave packs 512 products x*g (and squares) into a 16x32 f16 A-matrix,
// multiplies by all-ones B with v_wmma_f32_16x16x32_f16 accumulating into a
// f32 C tile.  The gate segment for each wave is staged into a private LDS
// slot with GLOBAL_LOAD_ASYNC_TO_LDS_B128 (ASYNCcnt) when available, so the
// async engine fills LDS while x streams from global.
// ---------------------------------------------------------------------------
__global__ void k_stats(const float* __restrict__ x, const float* __restrict__ g,
                        float* __restrict__ mu, float* __restrict__ rstd) {
    int bc = blockIdx.x;                 // 0..2047
    int b  = bc >> 8;
    const float* xp = x + (size_t)bc * NHW;
    const float* gp = g + (size_t)b  * NHW;
    int wave = threadIdx.x >> 5;         // 0..7
    int lane = threadIdx.x & 31;

#if USE_ASYNC_LDS
    __shared__ float sg[8 * 512];        // 2KB private slot per wave
    float* slot = sg + wave * 512;
#endif

    v8f  csum = {};
    v8f  csq  = {};
    v16h ones;
#pragma unroll
    for (int j = 0; j < 16; ++j) ones[j] = (_Float16)1.0f;

#pragma unroll
    for (int it = 0; it < 4; ++it) {
        int off = wave * 2048 + it * 512 + lane * 16;
        const float4* px = (const float4*)(xp + off);

#if USE_ASYNC_LDS
        // Wave-synchronous async stage of this wave's 512-float g segment.
#pragma unroll
        for (int q = 0; q < 4; ++q) {
            __builtin_amdgcn_global_load_async_to_lds_b128(
                AS_GLOBAL_V4I(gp + off + q * 4),
                AS_LDS_V4I(slot + lane * 16 + q * 4),
                0, 0);
        }
        __builtin_amdgcn_s_wait_asynccnt(0);
        const float4* pg = (const float4*)(slot + lane * 16);
#else
        const float4* pg = (const float4*)(gp + off);
#endif

        v16h av, aq;
#pragma unroll
        for (int q = 0; q < 4; ++q) {
            float4 xv = px[q];
            float4 gv = pg[q];
            float v0 = xv.x * gv.x, v1 = xv.y * gv.y;
            float v2 = xv.z * gv.z, v3 = xv.w * gv.w;
            av[q * 4 + 0] = (_Float16)v0;        aq[q * 4 + 0] = (_Float16)(v0 * v0);
            av[q * 4 + 1] = (_Float16)v1;        aq[q * 4 + 1] = (_Float16)(v1 * v1);
            av[q * 4 + 2] = (_Float16)v2;        aq[q * 4 + 2] = (_Float16)(v2 * v2);
            av[q * 4 + 3] = (_Float16)v3;        aq[q * 4 + 3] = (_Float16)(v3 * v3);
        }
        csum = __builtin_amdgcn_wmma_f32_16x16x32_f16(false, av, false, ones,
                                                      (short)0, csum, false, false);
        csq  = __builtin_amdgcn_wmma_f32_16x16x32_f16(false, aq, false, ones,
                                                      (short)0, csq,  false, false);
    }

    float s = csum[0] + csum[1] + csum[2] + csum[3] +
              csum[4] + csum[5] + csum[6] + csum[7];
    float q = csq[0] + csq[1] + csq[2] + csq[3] +
              csq[4] + csq[5] + csq[6] + csq[7];
    s += __shfl_xor(s, 16);              // lanes 0-15 hold M=0..7, 16-31 hold M=8..15
    q += __shfl_xor(q, 16);

    __shared__ float ls[8], lq[8];
    if (lane == 0) { ls[wave] = s; lq[wave] = q; }
    __syncthreads();
    if (threadIdx.x == 0) {
        float S = 0.f, Q = 0.f;
#pragma unroll
        for (int i = 0; i < 8; ++i) { S += ls[i]; Q += lq[i]; }
        float muv = S * (1.f / (float)NHW);
        float var = Q * (1.f / (float)NHW) - muv * muv;
        var = fmaxf(var, 0.f);
        mu[bc]   = muv;
        rstd[bc] = rsqrtf(var + EPSV);
    }
}

// ---------------------------------------------------------------------------
// K5: out = relu(5 * ((x*g - mu)*rstd*gamma + beta)), float4-vectorized.
// ---------------------------------------------------------------------------
__global__ void k_final(const float* __restrict__ x, const float* __restrict__ g,
                        const float* __restrict__ mu, const float* __restrict__ rstd,
                        const float* __restrict__ gamma, const float* __restrict__ beta,
                        float* __restrict__ out) {
    size_t i4 = (size_t)blockIdx.x * blockDim.x + threadIdx.x;  // exact coverage
    size_t i  = i4 * 4;
    size_t bc = i / NHW;
    int    p  = (int)(i % NHW);
    int    b  = (int)(bc >> 8);
    int    c  = (int)(bc & 255);
    float4 xv = *(const float4*)(x + i);
    float4 gv = *(const float4*)(g + (size_t)b * NHW + p);
    float muv = mu[bc], rs = rstd[bc], ga = gamma[c], be = beta[c];
    float4 o;
    o.x = fmaxf(0.f, SCL * (((xv.x * gv.x) - muv) * rs * ga + be));
    o.y = fmaxf(0.f, SCL * (((xv.y * gv.y) - muv) * rs * ga + be));
    o.z = fmaxf(0.f, SCL * (((xv.z * gv.z) - muv) * rs * ga + be));
    o.w = fmaxf(0.f, SCL * (((xv.w * gv.w) - muv) * rs * ga + be));
    *(float4*)(out + i) = o;
}

// ---------------------------------------------------------------------------
extern "C" void kernel_launch(void* const* d_in, const int* in_sizes, int n_in,
                              void* d_out, int out_size, void* d_ws, size_t ws_size,
                              hipStream_t stream) {
    const float* x      = (const float*)d_in[0];   // [8,256,128,128]
    const float* masks  = (const float*)d_in[1];   // [8,512,512]
    const float* conv_w = (const float*)d_in[2];   // [1,3,7,7] = 147
    const float* gamma  = (const float*)d_in[3];   // [256]
    const float* beta   = (const float*)d_in[4];   // [256]
    float* out = (float*)d_out;

    float* ws   = (float*)d_ws;
    float* m    = ws;                       // 131072
    float* avg  = ws + 131072;              // 131072
    float* mx   = ws + 262144;              // 131072
    float* g    = ws + 393216;              // 131072
    float* muv  = ws + 524288;              // 2048
    float* rstd = ws + 526336;              // 2048

    (void)in_sizes; (void)n_in; (void)out_size; (void)ws_size;

    k_resize    <<<512,   256, 0, stream>>>(masks, m);
    k_chanreduce<<<512,   256, 0, stream>>>(x, m, avg, mx);
    k_conv_gate <<<512,   256, 0, stream>>>(avg, mx, m, conv_w, g);
    k_stats     <<<2048,  256, 0, stream>>>(x, g, muv, rstd);
    k_final     <<<32768, 256, 0, stream>>>(x, g, muv, rstd, gamma, beta, out);
}
